// MultiHeadSINDyAttention_49349174231129
// MI455X (gfx1250) — matrix operand order repaired and checked by reading, not compiled, not run on hardware
//
#include <hip/hip_runtime.h>
#include <hip/hip_bf16.h>

typedef __attribute__((ext_vector_type(16))) _Float16 v16h;
typedef __attribute__((ext_vector_type(8)))  float    v8f;
typedef __attribute__((ext_vector_type(4)))  int      v4i;
typedef __attribute__((ext_vector_type(4)))  unsigned int v4u;

static constexpr int N_  = 4;
static constexpr int L_  = 2048;
static constexpr int EQ  = 512;
static constexpr int ET  = 512;
static constexpr int H_  = 8;
static constexpr int EH  = 64;
static constexpr int FC  = 8;
static constexpr float NEGINF = -1e30f;

#define NTHREADS 256
#define WPB 8   // waves per block (wave32)

// ---- async-to-LDS path (CDNA5) ---------------------------------------------
#if defined(__gfx1250__) && __has_builtin(__builtin_amdgcn_global_load_async_to_lds_b128)
#define HAVE_ASYNC 1
#else
#define HAVE_ASYNC 0
#endif

#if HAVE_ASYNC
typedef __attribute__((address_space(1))) v4i gv4i;   // global int4
typedef __attribute__((address_space(3))) v4i lv4i;   // LDS int4
__device__ __forceinline__ void async_copy16(const void* g, void* l) {
  gv4i* gp = (gv4i*)(unsigned long long)(uintptr_t)g;
  lv4i* lp = (lv4i*)(unsigned int)(uintptr_t)l;   // LDS offset = addr[31:0]
  __builtin_amdgcn_global_load_async_to_lds_b128(gp, lp, 0, 0);
}
#if __has_builtin(__builtin_amdgcn_s_wait_asynccnt)
#define WAIT_ASYNC(n) __builtin_amdgcn_s_wait_asynccnt(n)
#else
#define WAIT_ASYNC(n) asm volatile("s_wait_asynccnt %0" ::"i"(n) : "memory")
#endif
#else
__device__ __forceinline__ void async_copy16(const void* g, void* l) {
  *(v4u*)l = *(const v4u*)g;      // synchronous fallback
}
#define WAIT_ASYNC(n)
#endif

__device__ __forceinline__ int lane_id() { return threadIdx.x & 31; }

// ---- fragment helpers (contiguous k = 16*(lane/16) + elem, wave32) -------------------

// A-matrix 16x32 f16 (MxK), row-major source with leading dim ld.
__device__ __forceinline__ v16h load_a_frag(const _Float16* p, int ld) {
  const int l  = lane_id();
  const int m  = l & 15;
  const int kb = (l < 16) ? 0 : 16;
  const _Float16* row = p + (size_t)m * ld;
  v16h a;
#pragma unroll
  for (int i = 0; i < 8; ++i) {
    const int kk = kb + 2 * i;
    a[2 * i]     = row[kk];
    a[2 * i + 1] = row[kk + 1];
  }
  return a;
}

// Packed-B layout: for a KxN matrix, tile (kt = k/32, nt2 = n/16):
//   lane = ((k&31)>>4)*16 + (n&15),  elem = k&15  (elem == local k16!)
// so each lane's 16 halves are CONTIGUOUS -> single 32B v16h load.
__device__ __forceinline__ size_t pack_b_index(int k, int n, int N) {
  const int kt = k >> 5, kl = k & 31;
  return (((size_t)kt * (N >> 4) + (n >> 4)) * 32 + ((kl >> 4) * 16 + (n & 15))) * 16 +
         (kl & 15);
}

__device__ __forceinline__ v16h load_b_packed(const _Float16* packed, int kt,
                                              int nt2, int ntn) {
  const v16h* p = (const v16h*)packed;
  return p[((size_t)kt * ntn + nt2) * 32 + lane_id()];
}

// B fragment where logical B = S^T for row-major S (NxK): contiguous K pairs.
__device__ __forceinline__ v16h load_bT_frag(const _Float16* p, int ld) {
  const int l  = lane_id();
  const int n  = l & 15;
  const int kb = (l < 16) ? 0 : 16;
  const _Float16* row = p + (size_t)n * ld;
  v16h b;
#pragma unroll
  for (int i = 0; i < 8; ++i) {
    b[2 * i]     = row[kb + 2 * i];
    b[2 * i + 1] = row[kb + 2 * i + 1];
  }
  return b;
}

__device__ __forceinline__ v8f zero8() {
  v8f z = {0.f, 0.f, 0.f, 0.f, 0.f, 0.f, 0.f, 0.f};
  return z;
}

__device__ __forceinline__ v8f wmma_f16(v16h a, v16h b, v8f c) {
  return __builtin_amdgcn_wmma_f32_16x16x32_f16(false, a, false, b, (short)0, c,
                                                false, false);
}

// ---- small prep kernels ----------------------------------------------------

__global__ void cvt_f32_f16_kernel(const float* __restrict__ src,
                                   _Float16* __restrict__ dst, int n) {
  int i = blockIdx.x * blockDim.x + threadIdx.x;
  const int stride = gridDim.x * blockDim.x;
  for (; i < n; i += stride) dst[i] = (_Float16)src[i];
}

// f32 row-major KxN -> f16 packed-B layout
__global__ void pack_b_f32_kernel(const float* __restrict__ src,
                                  _Float16* __restrict__ dst, int K, int N) {
  int i = blockIdx.x * blockDim.x + threadIdx.x;
  const int total = K * N;
  const int stride = gridDim.x * blockDim.x;
  for (; i < total; i += stride) {
    const int k = i / N, n = i % N;
    dst[pack_b_index(k, n, N)] = (_Float16)src[i];
  }
}

// M_h = I + (Xi_h - Xi_h^T)   (DT = 1.0), row-major f16 [H][64][64]
__global__ void build_m_kernel(const float* __restrict__ Xi,
                               _Float16* __restrict__ Mh) {
  int i = blockIdx.x * blockDim.x + threadIdx.x;
  const int total = H_ * EH * EH;
  const int stride = gridDim.x * blockDim.x;
  for (; i < total; i += stride) {
    const int h   = i / (EH * EH);
    const int rem = i % (EH * EH);
    const int r   = rem / EH;
    const int c   = rem % EH;
    float a = Xi[h * EH * EH + r * EH + c] - Xi[h * EH * EH + c * EH + r];
    if (r == c) a += 1.0f;
    Mh[i] = (_Float16)a;
  }
}

// ---- QKV projection ---------------------------------------------------------
// [8192x512] @ packed[512x1536] + bqkv.  Block: 8 waves share one 64-col panel.
// Q scaled by 1/sqrt(64); Q/K row-major [b,h,s,64]; V written packed-B per bh.
__global__ void qkv_gemm_kernel(const _Float16* __restrict__ X,
                                const _Float16* __restrict__ Wpk,
                                const float* __restrict__ bqkv,
                                _Float16* __restrict__ Qb,
                                _Float16* __restrict__ Kb,
                                _Float16* __restrict__ Vpk) {
  __shared__ __align__(32) _Float16 Bs[2][4 * 32 * 16];  // 2 x 4KB
  const int wave = threadIdx.x >> 5;
  const int tid  = threadIdx.x;
  const int mt   = blockIdx.x * WPB + wave;   // 0..511
  const int nt   = blockIdx.y;                // 0..23 (64-wide panels)
  const int ntn  = (3 * ET) >> 4;             // 96
  const int l  = lane_id();
  const int nn = l & 15;
  const int mb = (l < 16) ? 0 : 8;

  const _Float16* Arow = X + (size_t)mt * 16 * EQ;
  v8f acc[4] = {zero8(), zero8(), zero8(), zero8()};

  auto stage = [&](int buf, int kt) {
    const _Float16* src = Wpk + ((size_t)kt * ntn + nt * 4) * 32 * 16 + tid * 8;
    async_copy16(src, &Bs[buf][tid * 8]);     // 256 threads x 16B = 4KB panel
  };

  stage(0, 0);
  for (int kt = 0; kt < EQ / 32; ++kt) {
    const int cur = kt & 1;
    if (kt + 1 < EQ / 32) {
      stage(cur ^ 1, kt + 1);
      WAIT_ASYNC(1);
    } else {
      WAIT_ASYNC(0);
    }
    __syncthreads();
    const v16h a = load_a_frag(Arow + kt * 32, EQ);
    const v16h* bsv = (const v16h*)&Bs[cur][0];
#pragma unroll
    for (int f = 0; f < 4; ++f) acc[f] = wmma_f16(a, bsv[f * 32 + l], acc[f]);
    __syncthreads();
  }

#pragma unroll
  for (int f = 0; f < 4; ++f) {
    const int col = nt * 64 + f * 16 + nn;
    const int sec = col / ET;                 // 0=q 1=k 2=v
    const int e   = col % ET;
    const int h   = e >> 6;
    const int d   = e & 63;
    const float scale = (sec == 0) ? 0.125f : 1.0f;
    const float bias  = bqkv[col];
#pragma unroll
    for (int r = 0; r < 8; ++r) {
      const int row = mt * 16 + mb + r;
      const int bb  = row >> 11;
      const int s   = row & 2047;
      const float v = (acc[f][r] + bias) * scale;
      const size_t bh = (size_t)bb * H_ + h;
      if (sec == 0) {
        Qb[(bh * L_ + s) * EH + d] = (_Float16)v;
      } else if (sec == 1) {
        Kb[(bh * L_ + s) * EH + d] = (_Float16)v;
      } else {
        // packed-B over (k = s, n = d), per (b,h) slab of L*EH
        Vpk[bh * (L_ * EH) + pack_b_index(s, d, EH)] = (_Float16)v;
      }
    }
  }
}

// ---- causal flash attention -------------------------------------------------
__global__ void attn_kernel(const _Float16* __restrict__ Qb,
                            const _Float16* __restrict__ Kb,
                            const _Float16* __restrict__ Vpk,
                            _Float16* __restrict__ Attn) {
  __shared__ __align__(32) _Float16 Pst[WPB][16 * 32];  // P transpose staging
  const int wave = threadIdx.x >> 5;
  const int idx  = blockIdx.x * WPB + wave;
  const int qt   = idx & 127;
  const int bh   = idx >> 7;
  const int l  = lane_id();
  const int nn = l & 15;
  const int mb = (l < 16) ? 0 : 8;

  const size_t base = (size_t)bh * L_ * EH;
  const _Float16* Q  = Qb + base + (size_t)qt * 16 * EH;
  const _Float16* Kp = Kb + base;
  const _Float16* Vp = Vpk + base;            // packed-B layout per bh
  const int qi0 = qt * 16;

  const v16h qa0 = load_a_frag(Q, EH);
  const v16h qa1 = load_a_frag(Q + 32, EH);

  v8f acc[4] = {zero8(), zero8(), zero8(), zero8()};
  float mrow[8], lrow[8];
#pragma unroll
  for (int r = 0; r < 8; ++r) { mrow[r] = NEGINF; lrow[r] = 0.f; }
  _Float16* Pw = &Pst[wave][0];

  for (int j0 = 0; j0 < qi0 + 16; j0 += 32) {
    v8f s0 = zero8(), s1 = zero8();
    {
      const v16h b0 = load_bT_frag(Kp + (size_t)j0 * EH, EH);
      const v16h b1 = load_bT_frag(Kp + (size_t)j0 * EH + 32, EH);
      s0 = wmma_f16(qa0, b0, s0);
      s0 = wmma_f16(qa1, b1, s0);
    }
    {
      const v16h b0 = load_bT_frag(Kp + (size_t)(j0 + 16) * EH, EH);
      const v16h b1 = load_bT_frag(Kp + (size_t)(j0 + 16) * EH + 32, EH);
      s1 = wmma_f16(qa0, b0, s1);
      s1 = wmma_f16(qa1, b1, s1);
    }
#pragma unroll
    for (int r = 0; r < 8; ++r) {
      const int q = qi0 + mb + r;
      float v0 = (j0 + nn <= q) ? s0[r] : NEGINF;
      float v1 = (j0 + 16 + nn <= q) ? s1[r] : NEGINF;
      float mx = fmaxf(v0, v1);
#pragma unroll
      for (int off = 8; off >= 1; off >>= 1)
        mx = fmaxf(mx, __shfl_xor(mx, off, 32));
      const float mnew = fmaxf(mrow[r], mx);
      const float corr = __expf(mrow[r] - mnew);
      const float p0 = __expf(v0 - mnew);
      const float p1 = __expf(v1 - mnew);
      float rs = p0 + p1;
#pragma unroll
      for (int off = 8; off >= 1; off >>= 1) rs += __shfl_xor(rs, off, 32);
      lrow[r] = lrow[r] * corr + rs;
      mrow[r] = mnew;
#pragma unroll
      for (int f = 0; f < 4; ++f) acc[f][r] *= corr;
      Pw[(mb + r) * 32 + nn]      = (_Float16)p0;
      Pw[(mb + r) * 32 + 16 + nn] = (_Float16)p1;
    }
    asm volatile("s_wait_dscnt 0\n\ts_wait_storecnt 0" ::: "memory");
    const v16h pa = load_a_frag(Pw, 32);
    const int kt = j0 >> 5;
#pragma unroll
    for (int f = 0; f < 4; ++f) {
      const v16h bv = load_b_packed(Vp, kt, f, EH >> 4);
      acc[f] = wmma_f16(pa, bv, acc[f]);
    }
  }

  const int b = bh >> 3, h = bh & 7;
#pragma unroll
  for (int r = 0; r < 8; ++r) {
    const float inv = 1.0f / lrow[r];
    const int s = qi0 + mb + r;
#pragma unroll
    for (int f = 0; f < 4; ++f) {
      const float v = acc[f][r] * inv;
      Attn[((size_t)b * L_ + s) * ET + h * 64 + f * 16 + nn] = (_Float16)v;
    }
  }
}

// ---- forecast chain: Wout = blockdiag(M_h) @ Win (packed in, packed out) ---
__global__ void chain_kernel(const _Float16* __restrict__ Mh,
                             const _Float16* __restrict__ Win,
                             _Float16* __restrict__ Wout) {
  const int wave = threadIdx.x >> 5;
  const int idx  = blockIdx.x * WPB + wave;   // 0..255
  const int nt   = idx & 7;
  const int mt   = idx >> 3;
  const int h    = mt >> 2;
  const int r0   = (mt & 3) * 16;
  const int l  = lane_id();
  const int nn = l & 15;
  const int mb = (l < 16) ? 0 : 8;

  v8f acc[4] = {zero8(), zero8(), zero8(), zero8()};
  for (int k0 = 0; k0 < EH; k0 += 32) {
    const v16h a = load_a_frag(Mh + (size_t)h * EH * EH + (size_t)r0 * EH + k0, EH);
    const int ktg = h * 2 + (k0 >> 5);        // global K-tile (K = 512 rows)
#pragma unroll
    for (int f = 0; f < 4; ++f) {
      const v16h b = load_b_packed(Win, ktg, nt * 4 + f, ET >> 4);
      acc[f] = wmma_f16(a, b, acc[f]);
    }
  }
#pragma unroll
  for (int f = 0; f < 4; ++f)
#pragma unroll
    for (int r = 0; r < 8; ++r) {
      const int row = h * EH + r0 + mb + r;
      const int col = nt * 64 + f * 16 + nn;
      Wout[pack_b_index(row, col, ET)] = (_Float16)acc[f][r];
    }
}

// ---- fused forecast+output GEMM --------------------------------------------
// out[b, step, s, :] = Attn[b,s,:] @ Wn[step] + bo  (Wn packed; async-staged)
__global__ void out_gemm_kernel(const _Float16* __restrict__ Attn,
                                const _Float16* __restrict__ Wn,
                                const float* __restrict__ bo,
                                float* __restrict__ Out) {
  __shared__ __align__(32) _Float16 Bs[2][4 * 32 * 16];
  const int step = blockIdx.z;
  const int nt   = blockIdx.y;                // 0..7
  const int wave = threadIdx.x >> 5;
  const int tid  = threadIdx.x;
  const int mt   = blockIdx.x * WPB + wave;   // 0..511
  const _Float16* Wp = Wn + (size_t)step * ET * EQ;
  const int ntn = EQ >> 4;                    // 32
  const int l  = lane_id();
  const int nn = l & 15;
  const int mb = (l < 16) ? 0 : 8;

  const _Float16* Arow = Attn + (size_t)mt * 16 * ET;
  v8f acc[4] = {zero8(), zero8(), zero8(), zero8()};

  auto stage = [&](int buf, int kt) {
    const _Float16* src = Wp + ((size_t)kt * ntn + nt * 4) * 32 * 16 + tid * 8;
    async_copy16(src, &Bs[buf][tid * 8]);
  };

  stage(0, 0);
  for (int kt = 0; kt < ET / 32; ++kt) {
    const int cur = kt & 1;
    if (kt + 1 < ET / 32) {
      stage(cur ^ 1, kt + 1);
      WAIT_ASYNC(1);
    } else {
      WAIT_ASYNC(0);
    }
    __syncthreads();
    const v16h a = load_a_frag(Arow + kt * 32, ET);
    const v16h* bsv = (const v16h*)&Bs[cur][0];
#pragma unroll
    for (int f = 0; f < 4; ++f) acc[f] = wmma_f16(a, bsv[f * 32 + l], acc[f]);
    __syncthreads();
  }

#pragma unroll
  for (int f = 0; f < 4; ++f) {
    const int col  = nt * 64 + f * 16 + nn;
    const float bias = bo[col];
#pragma unroll
    for (int r = 0; r < 8; ++r) {
      const int row = mt * 16 + mb + r;
      const int bb  = row >> 11;
      const int s   = row & 2047;
      Out[(((size_t)bb * FC + step) * L_ + s) * EQ + col] = acc[f][r] + bias;
    }
  }
}

// ---- launcher ---------------------------------------------------------------

extern "C" void kernel_launch(void* const* d_in, const int* in_sizes, int n_in,
                              void* d_out, int out_size, void* d_ws, size_t ws_size,
                              hipStream_t stream) {
  const float* query = (const float*)d_in[0];
  const float* Wqkv  = (const float*)d_in[3];
  const float* bqkv  = (const float*)d_in[4];
  const float* Wo    = (const float*)d_in[5];
  const float* bo    = (const float*)d_in[6];
  const float* Xi    = (const float*)d_in[7];
  float* out = (float*)d_out;

  // workspace carve-up (f16 elements)
  _Float16* X16  = (_Float16*)d_ws;                   // 8192*512
  _Float16* Wqpk = X16  + (size_t)8192 * 512;         // 512*1536 packed
  _Float16* Qb   = Wqpk + (size_t)512 * 1536;         // [b,h,s,64]
  _Float16* Kb   = Qb   + (size_t)N_ * H_ * L_ * EH;
  _Float16* Vpk  = Kb   + (size_t)N_ * H_ * L_ * EH;  // packed per bh
  _Float16* At16 = Vpk  + (size_t)N_ * H_ * L_ * EH;  // 8192*512
  _Float16* Mh   = At16 + (size_t)8192 * 512;         // 8*64*64
  _Float16* Wopk = Mh   + (size_t)H_ * EH * EH;       // 512*512 packed
  _Float16* Wn   = Wopk + (size_t)ET * EQ;            // 8*512*512 packed

  // 1) converts / packs
  cvt_f32_f16_kernel<<<1024, NTHREADS, 0, stream>>>(query, X16, 8192 * 512);
  pack_b_f32_kernel<<<768, NTHREADS, 0, stream>>>(Wqkv, Wqpk, 512, 1536);
  pack_b_f32_kernel<<<256, NTHREADS, 0, stream>>>(Wo, Wopk, 512, 512);
  build_m_kernel<<<128, NTHREADS, 0, stream>>>(Xi, Mh);

  // 2) QKV projection: grid (512/8 mt-groups, 24 col panels)
  qkv_gemm_kernel<<<dim3(64, 24), NTHREADS, 0, stream>>>(X16, Wqpk, bqkv, Qb,
                                                         Kb, Vpk);

  // 3) causal flash attention
  attn_kernel<<<512, NTHREADS, 0, stream>>>(Qb, Kb, Vpk, At16);

  // 4) forecast chain: Wn[n] = M^(n+1) @ Wo (packed in/out)
  for (int n = 0; n < FC; ++n) {
    const _Float16* win = (n == 0) ? Wopk : (Wn + (size_t)(n - 1) * ET * EQ);
    chain_kernel<<<32, NTHREADS, 0, stream>>>(Mh, win, Wn + (size_t)n * ET * EQ);
  }

  // 5) fused forecast+output GEMMs
  out_gemm_kernel<<<dim3(64, 8, FC), NTHREADS, 0, stream>>>(At16, Wn, bo, out);
}
